// MambaLayers_7413113553058
// MI455X (gfx1250) — compile-verified, matrix operand order checked by reading, not run on hardware
//
#include <hip/hip_runtime.h>
#include <hip/hip_bf16.h>
#include <math.h>

// Problem constants (match reference)
#define BB 4
#define LL 2048
#define DD 512
#define NLAYER 2
#define EDIM 1024
#define NSTATE 16
#define RRANK 32
#define KCONV 4
#define BL (BB*LL)               // 8192 rows
#define DBCW (RRANK + 2*NSTATE)  // 64

typedef __attribute__((ext_vector_type(16))) _Float16 v16h;
typedef __attribute__((ext_vector_type(8)))  float    v8f;
typedef __attribute__((ext_vector_type(4)))  unsigned int u32x4;

union AV { v16h h; u32x4 u[2]; };

__device__ __forceinline__ float sigmoidf_(float x) { return 1.0f / (1.0f + __expf(-x)); }

// ---------------- f32 -> f16 conversion (weights, small) ----------------
__global__ void cvt_f16(const float* __restrict__ src, _Float16* __restrict__ dst, int n) {
  int i = blockIdx.x * blockDim.x + threadIdx.x;
  if (i < n) dst[i] = (_Float16)src[i];
}

// ---------------- mask * x, RMSNorm, emit f16 normed activations ----------------
// one block per row (b*L+t), 256 threads, 2 elems/thread over D=512
__global__ void mask_rmsnorm(float* __restrict__ xres, const int* __restrict__ mask,
                             const float* __restrict__ nw, _Float16* __restrict__ xn) {
  int row = blockIdx.x;
  int tid = threadIdx.x;
  float m = (float)mask[row];
  __shared__ float red[256];
  size_t base = (size_t)row * DD;
  float v0 = xres[base + tid]       * m;
  float v1 = xres[base + tid + 256] * m;
  xres[base + tid]       = v0;
  xres[base + tid + 256] = v1;
  red[tid] = v0 * v0 + v1 * v1;
  __syncthreads();
  for (int s = 128; s > 0; s >>= 1) {
    if (tid < s) red[tid] += red[tid + s];
    __syncthreads();
  }
  float inv = rsqrtf(red[0] / (float)DD + 1e-5f);
  xn[base + tid]       = (_Float16)(v0 * inv * nw[tid]);
  xn[base + tid + 256] = (_Float16)(v1 * inv * nw[tid + 256]);
}

// ---------------- WMMA GEMM: C[M,N] = A[M,K] * W[N,K]^T  (f16 in, f32 acc) ----
// Register-blocked: each wave owns a 32x64 tile (2 M-subtiles x 4 N-subtiles,
// 8 WMMA accumulators). Block = 4 waves stacked in M -> 128x64 per block.
// EPI 0: store; EPI 1: softplus(c + bias[col]); EPI 2: Cout[idx] += c (residual)
template<int EPI>
__global__ void wmma_gemm(const _Float16* __restrict__ A, const _Float16* __restrict__ Bw,
                          float* __restrict__ Cout, const float* __restrict__ bias,
                          int Kdim, int Ncols) {
  int wave = threadIdx.x >> 5;
  int lane = threadIdx.x & 31;
  int tileM = blockIdx.y * 128 + wave * 32;
  int tileN = blockIdx.x * 64;
  int lh = lane & 15;   // A row / B^T row within 16-wide subtile
  int kg = lane >> 4;   // K-group select per 16-bit WMMA operand layout

  const _Float16* Ar[2];
  Ar[0] = A + (size_t)(tileM + lh) * Kdim;
  Ar[1] = Ar[0] + (size_t)16 * Kdim;
  const _Float16* Br[4];
#pragma unroll
  for (int nt = 0; nt < 4; ++nt)
    Br[nt] = Bw + (size_t)(tileN + nt * 16 + lh) * Kdim;

  v8f c[2][4];
#pragma unroll
  for (int mi = 0; mi < 2; ++mi)
#pragma unroll
    for (int nt = 0; nt < 4; ++nt)
      c[mi][nt] = (v8f){0.f, 0.f, 0.f, 0.f, 0.f, 0.f, 0.f, 0.f};

  for (int k0 = 0; k0 < Kdim; k0 += 32) {
    int o0 = k0 + kg * 8;   // lanes 0-15: K 0..7 / 16..23 ; lanes 16-31: 8..15 / 24..31
    AV a[2], b[4];
#pragma unroll
    for (int mi = 0; mi < 2; ++mi) {
      a[mi].u[0] = *(const u32x4*)(Ar[mi] + o0);
      a[mi].u[1] = *(const u32x4*)(Ar[mi] + o0 + 16);
      __builtin_prefetch(Ar[mi] + o0 + 32, 0, 0);   // global_prefetch_b8
    }
#pragma unroll
    for (int nt = 0; nt < 4; ++nt) {
      b[nt].u[0] = *(const u32x4*)(Br[nt] + o0);
      b[nt].u[1] = *(const u32x4*)(Br[nt] + o0 + 16);
      __builtin_prefetch(Br[nt] + o0 + 32, 0, 0);
    }
#pragma unroll
    for (int mi = 0; mi < 2; ++mi)
#pragma unroll
      for (int nt = 0; nt < 4; ++nt)
        c[mi][nt] = __builtin_amdgcn_wmma_f32_16x16x32_f16(
            false, a[mi].h, false, b[nt].h, (short)0, c[mi][nt], false, false);
  }

  // C/D layout: VGPR j -> M = j + 8*(lane>=16), N = lane&15
#pragma unroll
  for (int mi = 0; mi < 2; ++mi) {
    int rbase = tileM + mi * 16 + kg * 8;
#pragma unroll
    for (int nt = 0; nt < 4; ++nt) {
      int col = tileN + nt * 16 + lh;
#pragma unroll
      for (int j = 0; j < 8; ++j) {
        float v = c[mi][nt][j];
        size_t idx = (size_t)(rbase + j) * Ncols + col;
        if (EPI == 1) {
          v += bias[col];
          v = (v > 20.f) ? v : log1pf(__expf(v));   // softplus
          Cout[idx] = v;
        } else if (EPI == 2) {
          Cout[idx] += v;                            // residual add into x
        } else {
          Cout[idx] = v;
        }
      }
    }
  }
}

// ---------------- depthwise causal conv (K=4) + bias + SiLU -------------------
__global__ void conv_silu(const float* __restrict__ xz, const float* __restrict__ cw,
                          const float* __restrict__ cb, float* __restrict__ xc32,
                          _Float16* __restrict__ xc16) {
  size_t gid = (size_t)blockIdx.x * blockDim.x + threadIdx.x;   // over BL*ED
  int e = (int)(gid % EDIM);
  size_t bt = gid / EDIM;                // b*L + t
  int t = (int)(bt % LL);
  size_t b0 = bt - t;                    // b*L
  float acc = cb[e];
#pragma unroll
  for (int k = 0; k < KCONV; ++k) {
    int tt = t - (KCONV - 1) + k;
    if (tt >= 0) acc += cw[e * KCONV + k] * xz[(b0 + tt) * (size_t)(2 * EDIM) + e];
  }
  float s = acc * sigmoidf_(acc);
  xc32[gid] = s;
  xc16[gid] = (_Float16)s;
}

// ---------------- dt slice of dbc -> f16 operand for delta GEMM ---------------
__global__ void dt_cvt(const float* __restrict__ dbc, _Float16* __restrict__ dt16) {
  int i = blockIdx.x * blockDim.x + threadIdx.x;   // over BL*R
  int r = i % RRANK;
  int mrow = i / RRANK;
  dt16[i] = (_Float16)dbc[(size_t)mrow * DBCW + r];
}

// ---------------- selective scan + D skip + SiLU(z) gate ----------------------
// thread = one (b, e) channel; h[16] in registers; B_t/C_t broadcast by shuffle
__global__ void ssm_scan(const float* __restrict__ delta, const float* __restrict__ xc32,
                         const float* __restrict__ dbc, const float* __restrict__ xz,
                         const float* __restrict__ A_log, const float* __restrict__ Dp,
                         _Float16* __restrict__ yg) {
  int e = blockIdx.x * blockDim.x + threadIdx.x;
  int b = blockIdx.y;
  int lane = threadIdx.x & 31;
  float a[NSTATE];
#pragma unroll
  for (int n = 0; n < NSTATE; ++n) a[n] = -__expf(A_log[e * NSTATE + n]);
  float de = Dp[e];
  float h[NSTATE];
#pragma unroll
  for (int n = 0; n < NSTATE; ++n) h[n] = 0.f;
  size_t base = (size_t)b * LL;
  for (int t = 0; t < LL; ++t) {
    size_t r = base + t;
    float dlt = delta[r * EDIM + e];
    float xv  = xc32[r * EDIM + e];
    float zv  = xz[r * (size_t)(2 * EDIM) + EDIM + e];
    // lanes 0-15 fetch B_t[lane], lanes 16-31 fetch C_t[lane-16]
    float bc  = dbc[r * DBCW + RRANK + lane];
    float dx = dlt * xv;
    float y = 0.f;
#pragma unroll
    for (int n = 0; n < NSTATE; ++n) {
      float Bn = __shfl(bc, n, 32);
      float Cn = __shfl(bc, n + NSTATE, 32);
      float dA = __expf(dlt * a[n]);
      h[n] = dA * h[n] + dx * Bn;
      y = fmaf(h[n], Cn, y);
    }
    y += de * xv;
    float g = zv * sigmoidf_(zv);
    yg[r * EDIM + e] = (_Float16)(y * g);
  }
}

// ---------------- trailing mask, output copy ----------------------------------
__global__ void apply_mask(float* __restrict__ xres, const int* __restrict__ mask) {
  size_t gid = (size_t)blockIdx.x * blockDim.x + threadIdx.x;   // over BL*D
  int row = (int)(gid / DD);
  xres[gid] *= (float)mask[row];
}

__global__ void write_out(const float* __restrict__ xres, float* __restrict__ out) {
  size_t gid = (size_t)blockIdx.x * blockDim.x + threadIdx.x;
  size_t tot = (size_t)BL * DD;
  if (gid < tot) {
    out[gid] = xres[gid];
  } else if (gid < tot + (size_t)BB * DD) {
    size_t k = gid - tot;
    int b = (int)(k / DD);
    int col = (int)(k % DD);
    out[tot + k] = xres[(size_t)b * LL * DD + col];   // x[:,0,:]
  }
}

extern "C" void kernel_launch(void* const* d_in, const int* in_sizes, int n_in,
                              void* d_out, int out_size, void* d_ws, size_t ws_size,
                              hipStream_t stream) {
  (void)in_sizes; (void)n_in; (void)out_size; (void)ws_size;
  const float* x        = (const float*)d_in[0];
  const int*   mask     = (const int*)  d_in[1];
  const float* norm_w   = (const float*)d_in[2];
  const float* in_w     = (const float*)d_in[3];
  const float* conv_w   = (const float*)d_in[4];
  const float* conv_b   = (const float*)d_in[5];
  const float* xproj_w  = (const float*)d_in[6];
  const float* dtproj_w = (const float*)d_in[7];
  const float* dtproj_b = (const float*)d_in[8];
  const float* A_log    = (const float*)d_in[9];
  const float* Dp       = (const float*)d_in[10];
  const float* out_w    = (const float*)d_in[11];
  float* out = (float*)d_out;

  // carve scratch
  char* wsp = (char*)d_ws;
  auto carve = [&](size_t bytes) -> char* {
    char* p = wsp;
    wsp += (bytes + 255) & ~(size_t)255;
    return p;
  };
  float*    xres   = (float*)   carve((size_t)BL * DD * 4);
  _Float16* xn16   = (_Float16*)carve((size_t)BL * DD * 2);
  float*    xz     = (float*)   carve((size_t)BL * 2 * EDIM * 4);
  float*    xc32   = (float*)   carve((size_t)BL * EDIM * 4);
  _Float16* xc16   = (_Float16*)carve((size_t)BL * EDIM * 2);
  float*    dbc    = (float*)   carve((size_t)BL * DBCW * 4);
  _Float16* dt16   = (_Float16*)carve((size_t)BL * RRANK * 2);
  float*    delta  = (float*)   carve((size_t)BL * EDIM * 4);
  _Float16* yg16   = (_Float16*)carve((size_t)BL * EDIM * 2);
  _Float16* inw16  = (_Float16*)carve((size_t)2 * EDIM * DD * 2);
  _Float16* xprj16 = (_Float16*)carve((size_t)DBCW * EDIM * 2);
  _Float16* dtpw16 = (_Float16*)carve((size_t)EDIM * RRANK * 2);
  _Float16* outw16 = (_Float16*)carve((size_t)DD * EDIM * 2);

  hipMemcpyAsync(xres, x, (size_t)BL * DD * 4, hipMemcpyDeviceToDevice, stream);

  for (int i = 0; i < NLAYER; ++i) {
    // per-layer weight slices -> f16
    { int n = 2 * EDIM * DD;
      cvt_f16<<<(n + 255) / 256, 256, 0, stream>>>(in_w + (size_t)i * n, inw16, n); }
    { int n = DBCW * EDIM;
      cvt_f16<<<(n + 255) / 256, 256, 0, stream>>>(xproj_w + (size_t)i * n, xprj16, n); }
    { int n = EDIM * RRANK;
      cvt_f16<<<(n + 255) / 256, 256, 0, stream>>>(dtproj_w + (size_t)i * n, dtpw16, n); }
    { int n = DD * EDIM;
      cvt_f16<<<(n + 255) / 256, 256, 0, stream>>>(out_w + (size_t)i * n, outw16, n); }

    // x *= m ; xn = rmsnorm(x) in f16
    mask_rmsnorm<<<BL, 256, 0, stream>>>(xres, mask, norm_w + (size_t)i * DD, xn16);

    // xz = xn @ in_w^T   [8192 x 2048, K=512]
    wmma_gemm<0><<<dim3(2 * EDIM / 64, BL / 128), 128, 0, stream>>>(
        xn16, inw16, xz, nullptr, DD, 2 * EDIM);

    // depthwise causal conv + SiLU
    conv_silu<<<(int)((size_t)BL * EDIM / 256), 256, 0, stream>>>(
        xz, conv_w + (size_t)i * EDIM * KCONV, conv_b + (size_t)i * EDIM, xc32, xc16);

    // dbc = xc @ xproj_w^T   [8192 x 64, K=1024]
    wmma_gemm<0><<<dim3(DBCW / 64, BL / 128), 128, 0, stream>>>(
        xc16, xprj16, dbc, nullptr, EDIM, DBCW);

    // dt -> f16
    dt_cvt<<<(BL * RRANK + 255) / 256, 256, 0, stream>>>(dbc, dt16);

    // delta = softplus(dt @ dtproj_w^T + b)   [8192 x 1024, K=32]
    wmma_gemm<1><<<dim3(EDIM / 64, BL / 128), 128, 0, stream>>>(
        dt16, dtpw16, delta, dtproj_b + (size_t)i * EDIM, RRANK, EDIM);

    // sequential SSM scan + skip + gate, emits f16 operand for out GEMM
    ssm_scan<<<dim3(EDIM / 256, BB), 256, 0, stream>>>(
        delta, xc32, dbc, xz, A_log + (size_t)i * EDIM * NSTATE, Dp + (size_t)i * EDIM, yg16);

    // x += y @ out_w^T   [8192 x 512, K=1024]  (residual epilogue)
    wmma_gemm<2><<<dim3(DD / 64, BL / 128), 128, 0, stream>>>(
        yg16, outw16, xres, nullptr, EDIM, DD);

    // x *= m
    apply_mask<<<(int)((size_t)BL * DD / 256), 256, 0, stream>>>(xres, mask);
  }

  size_t tot = (size_t)BL * DD + (size_t)BB * DD;
  write_out<<<(int)((tot + 255) / 256), 256, 0, stream>>>(xres, out);
}
